// UserModel_13280038880097
// MI455X (gfx1250) — matrix operand order repaired and checked by reading, not compiled, not run on hardware
//
#include <hip/hip_runtime.h>
#include <hip/hip_bf16.h>

#define B_   32
#define S_   200
#define V_   128
#define NUMC 10000

typedef __attribute__((ext_vector_type(16))) _Float16 v16h;
typedef __attribute__((ext_vector_type(8)))  _Float16 half8;
typedef __attribute__((ext_vector_type(8)))  float    v8f;

// ---- WMMA operand loaders (CDNA5 16x16x32 f16 layouts, ISA 7.12.2 / 7.12.4) ----
// A (16x32, MxK): lane holds row M=lane%16; halves 0-7 = K {0..7 | 8..15}+base,
// halves 8-15 = +16 (lane group selects +8).
__device__ __forceinline__ v16h loadA16(const _Float16* row, int k0, int lane) {
    const _Float16* p = row + k0 + ((lane & 16) ? 8 : 0);
    v16h r;
    ((half8*)&r)[0] = *(const half8*)(p);
    ((half8*)&r)[1] = *(const half8*)(p + 16);
    return r;
}
// B (32x16, KxN): lane holds col N=lane%16; halves 0-7 = K {0..7 | 8..15}+base,
// halves 8-15 = +16 (lane group selects +8), same as A.  col points at W[n][0..127] (B=W^T).
__device__ __forceinline__ v16h loadB16(const _Float16* col, int k0, int lane) {
    const _Float16* p = col + k0 + ((lane & 16) ? 8 : 0);
    v16h r;
    ((half8*)&r)[0] = *(const half8*)(p);
    ((half8*)&r)[1] = *(const half8*)(p + 16);
    return r;
}

// ---------------- Kernel 0: precompute ----------------
// WhhF16, W1a16 casts; u_d = W_ih[:, :V]@v_d, u_r = W_ih[:, V:]@v_r (384 each);
// w_beta/w_d/w_r = W2a column-block @ vectors (128 each).
__global__ void prep_kernel(const float* __restrict__ W_hh, const float* __restrict__ W1a,
                            const float* __restrict__ W_ih, const float* __restrict__ v_d,
                            const float* __restrict__ v_r,  const float* __restrict__ W2a,
                            const float* __restrict__ v_beta,
                            _Float16* __restrict__ WhhF16, _Float16* __restrict__ W1a16,
                            float* __restrict__ u_d, float* __restrict__ u_r,
                            float* __restrict__ wbeta, float* __restrict__ wd,
                            float* __restrict__ wr) {
    int tid = threadIdx.x;  // 512 threads
    for (int i = tid; i < 3 * V_ * V_; i += 512) WhhF16[i] = (_Float16)W_hh[i];
    for (int i = tid; i < V_ * V_;     i += 512) W1a16[i]  = (_Float16)W1a[i];
    if (tid < 3 * V_) {
        float sd = 0.f, sr = 0.f;
        for (int k = 0; k < V_; ++k) {
            sd = fmaf(W_ih[tid * 2 * V_ + k],      v_d[k], sd);
            sr = fmaf(W_ih[tid * 2 * V_ + V_ + k], v_r[k], sr);
        }
        u_d[tid] = sd; u_r[tid] = sr;
    }
    if (tid < V_) {
        float sb = 0.f, sdd = 0.f, srr = 0.f;
        for (int k = 0; k < V_; ++k) {
            sb  = fmaf(W2a[tid * 3 * V_ + k],          v_beta[k], sb);
            sdd = fmaf(W2a[tid * 3 * V_ + V_ + k],     v_d[k],    sdd);
            srr = fmaf(W2a[tid * 3 * V_ + 2 * V_ + k], v_r[k],    srr);
        }
        wbeta[tid] = sb; wd[tid] = sdd; wr[tid] = srr;
    }
}

// ---------------- Kernel 1: GRU scan (WMMA recurrent GEMM) ----------------
// 2 blocks x 384 threads (12 waves). Block handles 16 batches. Per step:
// gh = h@W_hh.T + b_hh via 24 N-tiles of v_wmma_f32_16x16x32_f16 (2 tiles/wave),
// then gates with on-the-fly gx = gamma*u_d + r*u_r + b_ih.
__global__ void gru_kernel(const int* __restrict__ d_seq, const int* __restrict__ r_seq,
                           const float* __restrict__ D_emb,
                           const float* __restrict__ b_ih_g, const float* __restrict__ b_hh_g,
                           const _Float16* __restrict__ WhhF16,
                           const float* __restrict__ u_d, const float* __restrict__ u_r,
                           float* __restrict__ h_seq_out, _Float16* __restrict__ h16_out) {
    extern __shared__ char smem[];
    float*    sGh  = (float*)smem;              // 16*384
    float*    sH32 = sGh + 16 * 384;            // 16*128
    float*    sUd  = sH32 + 16 * 128;           // 384
    float*    sUr  = sUd + 384;                 // 384
    float*    sBih = sUr + 384;                 // 384
    float*    sBhh = sBih + 384;                // 384
    _Float16* sWhh = (_Float16*)(sBhh + 384);   // 384*128
    _Float16* sH16 = sWhh + 384 * 128;          // 16*128

    const int tid  = threadIdx.x;               // 0..383
    const int wave = tid >> 5, lane = tid & 31;
    const int bbase = blockIdx.x * 16;

    for (int i = tid; i < 3 * V_ * V_; i += 384) sWhh[i] = WhhF16[i];
    if (tid < 384) { sUd[tid] = u_d[tid]; sUr[tid] = u_r[tid];
                     sBih[tid] = b_ih_g[tid]; sBhh[tid] = b_hh_g[tid]; }
    for (int i = tid; i < 16 * V_; i += 384) { sH32[i] = 0.f; sH16[i] = (_Float16)0.f; }
    __syncthreads();

    const _Float16* arow = sH16 + (lane & 15) * V_;
    for (int t = 0; t < S_; ++t) {
        // ---- gh = h @ W_hh^T + b_hh ----
        #pragma unroll
        for (int q = 0; q < 2; ++q) {
            const int nt = wave + q * 12;
            const int n  = nt * 16 + (lane & 15);
            const _Float16* bcol = sWhh + n * V_;
            v8f c = {};
            #pragma unroll
            for (int k0 = 0; k0 < V_; k0 += 32) {
                v16h a = loadA16(arow, k0, lane);
                v16h b = loadB16(bcol, k0, lane);
                c = __builtin_amdgcn_wmma_f32_16x16x32_f16(
                        false, a, false, b, (short)0, c, false, false);
            }
            const int mh = (lane >> 4) * 8;
            const float bh = sBhh[n];
            #pragma unroll
            for (int j = 0; j < 8; ++j)
                sGh[(j + mh) * 384 + n] = c[j] + bh;
        }
        __syncthreads();
        // ---- gates ----
        for (int idx = tid; idx < 16 * V_; idx += 384) {
            const int bl = idx >> 7, j = idx & (V_ - 1);
            const int b  = bbase + bl;
            const float gamma = D_emb[d_seq[b * S_ + t]];
            const float rr    = (float)r_seq[b * S_ + t];
            const float gxr = fmaf(gamma, sUd[j],          fmaf(rr, sUr[j],          sBih[j]));
            const float gxz = fmaf(gamma, sUd[j + V_],     fmaf(rr, sUr[j + V_],     sBih[j + V_]));
            const float gxn = fmaf(gamma, sUd[j + 2 * V_], fmaf(rr, sUr[j + 2 * V_], sBih[j + 2 * V_]));
            const float* gh = sGh + bl * 384;
            const float rg = 1.f / (1.f + expf(-(gxr + gh[j])));
            const float zg = 1.f / (1.f + expf(-(gxz + gh[j + V_])));
            const float ng = tanhf(gxn + rg * gh[j + 2 * V_]);
            const float hn = (1.f - zg) * ng + zg * sH32[bl * V_ + j];
            sH32[bl * V_ + j] = hn;
            const _Float16 hh = (_Float16)hn;
            sH16[bl * V_ + j] = hh;
            h_seq_out[((size_t)b * S_ + t) * V_ + j] = hn;
            h16_out[((size_t)b * S_ + t) * V_ + j]   = hh;
        }
        __syncthreads();
    }
}

// ---------------- Kernel 2: alpha = relu(h@W1a.T + b1a)@W1b.T + b1b ----------------
// 100 blocks x 128 threads; 1 wave per 16-row M-tile (400 tiles = 6400 rows exact).
__global__ void alpha_kernel(const _Float16* __restrict__ h16seq,
                             const _Float16* __restrict__ w1a16,
                             const float* __restrict__ b1a, const float* __restrict__ W1b,
                             const float* __restrict__ b1b, float* __restrict__ alpha) {
    const int wave = threadIdx.x >> 5, lane = threadIdx.x & 31;
    const int mt = blockIdx.x * 4 + wave;
    const _Float16* arow = h16seq + (size_t)(mt * 16 + (lane & 15)) * V_;
    float acc[8];
    #pragma unroll
    for (int j = 0; j < 8; ++j) acc[j] = 0.f;
    for (int nt = 0; nt < 8; ++nt) {
        const int n = nt * 16 + (lane & 15);
        const _Float16* bcol = w1a16 + n * V_;
        v8f c = {};
        #pragma unroll
        for (int k0 = 0; k0 < V_; k0 += 32) {
            v16h a = loadA16(arow, k0, lane);
            v16h b = loadB16(bcol, k0, lane);
            c = __builtin_amdgcn_wmma_f32_16x16x32_f16(
                    false, a, false, b, (short)0, c, false, false);
        }
        const float bias = b1a[n], wv = W1b[n];
        #pragma unroll
        for (int j = 0; j < 8; ++j) {
            float y = c[j] + bias;
            y = y > 0.f ? y : 0.f;
            acc[j] = fmaf(y, wv, acc[j]);
        }
    }
    #pragma unroll
    for (int m = 8; m >= 1; m >>= 1)
        #pragma unroll
        for (int j = 0; j < 8; ++j) acc[j] += __shfl_xor(acc[j], m, 32);
    if ((lane & 15) == 0) {
        const int mh = (lane >> 4) * 8;
        #pragma unroll
        for (int j = 0; j < 8; ++j)
            alpha[mt * 16 + mh + j] = acc[j] + b1b[0];
    }
}

// ---------------- Kernel 3: counter scan (1 wave per batch) ----------------
__global__ void beta_kernel(const int* __restrict__ c_seq, const int* __restrict__ d_seq,
                            const int* __restrict__ r_seq, const float* __restrict__ D_emb,
                            const float* __restrict__ wbeta, const float* __restrict__ wd,
                            const float* __restrict__ wr, const float* __restrict__ b2a,
                            const float* __restrict__ W2b, const float* __restrict__ b2b,
                            float* __restrict__ Crow, float* __restrict__ newc) {
    const int b = threadIdx.x >> 5, lane = threadIdx.x & 31;
    float* C = Crow + (size_t)b * NUMC;
    for (int i = lane; i < NUMC; i += 32) C[i] = 0.f;
    const float bb = b2b[0];
    for (int t = 0; t < S_; ++t) {
        const int ct = c_seq[b * S_ + t];
        float beta = (lane == 0) ? C[ct] : 0.f;
        beta = __shfl(beta, 0, 32);
        const float gamma = D_emb[d_seq[b * S_ + t]];
        const float rr    = (float)r_seq[b * S_ + t];
        float p = 0.f;
        #pragma unroll
        for (int q = 0; q < 4; ++q) {
            const int j = lane + q * 32;
            float y = fmaf(beta, wbeta[j], fmaf(gamma, wd[j], fmaf(rr, wr[j], b2a[j])));
            y = fmaxf(y, 0.f);
            p = fmaf(y, W2b[j], p);
        }
        #pragma unroll
        for (int m = 16; m >= 1; m >>= 1) p += __shfl_xor(p, m, 32);
        const float nc = p + bb;
        if (lane == 0) { C[ct] = nc; newc[b * S_ + t] = nc; }
    }
}

// ---------------- Kernel 4: C_seq forward-fill (256 MB, b128 stores) ----------------
__global__ void fill_kernel(const int* __restrict__ c_seq, const float* __restrict__ newc,
                            float* __restrict__ outC) {
    __shared__ int   cs[S_];
    __shared__ float nc[S_];
    const int b = blockIdx.x / 10, blk = blockIdx.x % 10;
    const int tid = threadIdx.x;
    for (int i = tid; i < S_; i += 256) { cs[i] = c_seq[b * S_ + i]; nc[i] = newc[b * S_ + i]; }
    __syncthreads();
    const int c0 = (blk * 256 + tid) * 4;
    if (c0 >= NUMC) return;
    float4 cur{0.f, 0.f, 0.f, 0.f};
    for (int t = 0; t < S_; ++t) {
        const int u = cs[t] - c0;
        if ((unsigned)u < 4u) ((float*)&cur)[u] = nc[t];
        *(float4*)(outC + ((size_t)b * S_ + t) * NUMC + c0) = cur;
    }
}

extern "C" void kernel_launch(void* const* d_in, const int* in_sizes, int n_in,
                              void* d_out, int out_size, void* d_ws, size_t ws_size,
                              hipStream_t stream) {
    const int*   c_seq  = (const int*)d_in[0];
    const int*   d_seq  = (const int*)d_in[1];
    const int*   r_seq  = (const int*)d_in[2];
    const float* D_emb  = (const float*)d_in[3];
    const float* v_d    = (const float*)d_in[4];
    const float* v_r    = (const float*)d_in[5];
    const float* v_beta = (const float*)d_in[6];
    const float* W_ih   = (const float*)d_in[7];
    const float* W_hh   = (const float*)d_in[8];
    const float* b_ih   = (const float*)d_in[9];
    const float* b_hh   = (const float*)d_in[10];
    const float* W1a    = (const float*)d_in[11];
    const float* b1a    = (const float*)d_in[12];
    const float* W1b    = (const float*)d_in[13];
    const float* b1b    = (const float*)d_in[14];
    // d_in[15] = W2a, d_in[16] = b2a, d_in[17] = W2b, d_in[18] = b2b
    const float* W2a = (const float*)d_in[15];
    const float* b2a = (const float*)d_in[16];
    const float* W2b = (const float*)d_in[17];
    const float* b2b = (const float*)d_in[18];

    // workspace layout (bytes, 256-aligned)
    char* ws = (char*)d_ws;
    _Float16* WhhF16 = (_Float16*)(ws + 0);         //  98304 B
    _Float16* W1a16  = (_Float16*)(ws + 98304);     //  32768 B
    float*    vecs   = (float*)(ws + 131072);       //   4608 B
    float *u_d = vecs, *u_r = vecs + 384, *wbeta = vecs + 768,
          *wd = vecs + 896, *wr = vecs + 1024;
    _Float16* h16seq = (_Float16*)(ws + 135680);    // 1638400 B
    float*    newc   = (float*)(ws + 1774080);      //  25600 B
    float*    Crow   = (float*)(ws + 1799680);      // 1280000 B

    float* alpha_out = (float*)d_out;                       // [B,S]
    float* h_out     = alpha_out + B_ * S_;                 // [B,S,V]
    float* C_out     = h_out + (size_t)B_ * S_ * V_;        // [B,S,NUMC]

    prep_kernel<<<1, 512, 0, stream>>>(W_hh, W1a, W_ih, v_d, v_r, W2a, v_beta,
                                       WhhF16, W1a16, u_d, u_r, wbeta, wd, wr);

    const size_t gruLds = (16*384 + 16*128 + 4*384) * 4 + (384*128 + 16*128) * 2; // 141312
    gru_kernel<<<2, 384, gruLds, stream>>>(d_seq, r_seq, D_emb, b_ih, b_hh,
                                           WhhF16, u_d, u_r, h_out, h16seq);

    alpha_kernel<<<100, 128, 0, stream>>>(h16seq, W1a16, b1a, W1b, b1b, alpha_out);

    beta_kernel<<<1, 1024, 0, stream>>>(c_seq, d_seq, r_seq, D_emb,
                                        wbeta, wd, wr, b2a, W2b, b2b, Crow, newc);

    fill_kernel<<<B_ * 10, 256, 0, stream>>>(c_seq, newc, C_out);
}
